// TaskClassifier_4037269258286
// MI455X (gfx1250) — matrix operand order repaired and checked by reading, not compiled, not run on hardware
//
#include <hip/hip_runtime.h>
#include <hip/hip_bf16.h>
#include <math.h>

// ---------- problem constants (fixed by the reference setup) ----------
constexpr int Bn  = 16;       // batch
constexpr int Tn  = 11881;    // tokens (= 109^2)
constexpr int Pn  = 109;      // sqrt(T)
constexpr int Dn  = 384;      // feature dim (K)
constexpr int Cc  = 5;        // classes
constexpr int TOn = 128;      // descriptors per class
constexpr int Mn  = Cc * TOn; // 640 descriptor rows (GEMM M)

constexpr int MBLK  = 128;    // M rows per inner M-block (8 waves x 16)
constexpr int NBLK  = 64;     // t values per workgroup
constexpr int KSTEP = 32;     // K per WMMA
constexpr int NTILES = (Tn + NBLK - 1) / NBLK; // 186
constexpr int MLOOP  = Mn / MBLK;              // 5

typedef __attribute__((ext_vector_type(16))) _Float16 v16h;
typedef __attribute__((ext_vector_type(8)))  _Float16 v8h;
typedef __attribute__((ext_vector_type(4)))  _Float16 v4h;
typedef __attribute__((ext_vector_type(8)))  float    v8f;

union AFrag { v16h v; v8h h[2]; };

// Monotone packing: larger similarity wins; on equal sim, smaller t wins.
__device__ __forceinline__ unsigned long long pack_key(float v, int t) {
    unsigned u = __float_as_uint(v);
    u = (u & 0x80000000u) ? ~u : (u | 0x80000000u);   // sortable-float transform
    return ((unsigned long long)u << 32) | (unsigned)(0x7FFFFFFF - t);
}

__device__ __forceinline__ unsigned long long shfl_xor_u64(unsigned long long v, int m) {
    unsigned lo = (unsigned)v, hi = (unsigned)(v >> 32);
    lo = __shfl_xor(lo, m, 32);
    hi = __shfl_xor(hi, m, 32);
    return ((unsigned long long)hi << 32) | lo;
}

// ---------- kernel 0: zero the argmax table ----------
__global__ void init_best_kernel(unsigned long long* best, int n) {
    int i = blockIdx.x * blockDim.x + threadIdx.x;
    if (i < n) best[i] = 0ull;
}

// ---------- kernel 1: normalize desc rows -> f16 ----------
__global__ __launch_bounds__(128) void prep_desc_kernel(const float* __restrict__ desc,
                                                        _Float16* __restrict__ dn) {
    const int row = blockIdx.x;                 // 0..639  (= c*TO + o)
    const int tid = threadIdx.x;                // 0..127
    const float* src = desc + (size_t)row * Dn;
    __shared__ float part[4];

    float v0 = src[tid], v1 = src[tid + 128], v2 = src[tid + 256];
    float ss = v0 * v0 + v1 * v1 + v2 * v2;
    #pragma unroll
    for (int sh = 1; sh < 32; sh <<= 1) ss += __shfl_xor(ss, sh, 32);
    if ((tid & 31) == 0) part[tid >> 5] = ss;
    __syncthreads();
    float tot = part[0] + part[1] + part[2] + part[3];
    float inv = 1.0f / fmaxf(sqrtf(tot), 1e-8f);

    _Float16* dst = dn + (size_t)row * Dn;
    dst[tid]       = (_Float16)(v0 * inv);
    dst[tid + 128] = (_Float16)(v1 * inv);
    dst[tid + 256] = (_Float16)(v2 * inv);
}

// ---------- kernel 2: single-pass x tile + WMMA GEMM over ALL M + argmax ----------
// sim(m, t) = dot(dn[m], x[t]) * invnorm[t]  -- normalization deferred to epilogue,
// so raw x is converted to f16 in the same pass that accumulates sum-of-squares,
// and the LDS tile is reused for all 640 descriptor rows (x read once from HBM).
__global__ __launch_bounds__(256) void sim_argmax_kernel(const float* __restrict__ x,
                                                         const _Float16* __restrict__ dn,
                                                         unsigned long long* __restrict__ best) {
    __shared__ alignas(32) _Float16 xs[NBLK][Dn];   // 48KB raw-x f16 tile
    __shared__ float ssq[NBLK][4];
    __shared__ float invn[NBLK];

    const int tid  = threadIdx.x;
    const int lane = tid & 31;
    const int wave = tid >> 5;          // 0..7
    const int b    = blockIdx.y;        // batch
    const int t0   = blockIdx.x * NBLK; // token block

    // ---- stage 1: one pass over global x: f16 tile + sum-of-squares ----
    const int row = tid >> 2;           // 0..63
    const int seg = (tid & 3) * 96;     // 4 x 96 contiguous floats per row
    const int t   = t0 + row;
    const float* xr = x + ((size_t)b * Tn + (size_t)t) * Dn + seg;

    float ss = 0.0f;
    if (t < Tn) {
        #pragma unroll 4
        for (int i = 0; i < 96; i += 4) {
            float4 v = *(const float4*)(xr + i);
            ss += v.x * v.x + v.y * v.y + v.z * v.z + v.w * v.w;
            v4h h = { (_Float16)v.x, (_Float16)v.y, (_Float16)v.z, (_Float16)v.w };
            *(v4h*)&xs[row][seg + i] = h;
        }
    } else {
        v4h z = {};
        #pragma unroll 4
        for (int i = 0; i < 96; i += 4) *(v4h*)&xs[row][seg + i] = z;
    }
    ssq[row][tid & 3] = ss;
    __syncthreads();
    if ((tid & 3) == 0) {
        float tot = ssq[row][0] + ssq[row][1] + ssq[row][2] + ssq[row][3];
        invn[row] = 1.0f / fmaxf(sqrtf(tot), 1e-8f);
    }
    __syncthreads();

    // Per-lane fragment addressing (contiguous k = 16*(lane/16) + s for both A and B)
    const int akb  = (lane >> 4) * 16;  // A K sub-offset
    const int bcol = lane & 15;         // B N column = lane within half
    const int bkb  = (lane >> 4) * 16;  // B: lanes 16-31 hold K=16..31
    const int tl   = lane & 15;

    // invnorm for this lane's 4 token columns (uniform per N-subtile)
    const float in0 = invn[ 0 + tl];
    const float in1 = invn[16 + tl];
    const float in2 = invn[32 + tl];
    const float in3 = invn[48 + tl];

    // ---- stage 2: loop over all 5 M-blocks, reusing the LDS tile ----
    for (int mb = 0; mb < MLOOP; ++mb) {
        const int mw   = mb * MBLK + wave * 16;  // wave's 16 desc rows
        const int arow = mw + tl;                // lanes 0-15 & 16-31 both hold M=0..15
        const _Float16* abase = dn + (size_t)arow * Dn;
        __builtin_prefetch(abase, 0, 3);         // global_prefetch_b8 (dn is L2-resident)
        __builtin_prefetch(abase + Dn / 2, 0, 3);

        v8f acc0 = {}, acc1 = {}, acc2 = {}, acc3 = {};
        #pragma unroll 4
        for (int k0 = 0; k0 < Dn; k0 += KSTEP) {
            AFrag a;
            a.h[0] = *(const v8h*)(abase + k0 + akb);        // K = k0+akb .. +7
            a.h[1] = *(const v8h*)(abase + k0 + akb + 8);    // K = k0+akb+8 .. +15
            v16h bf0 = *(const v16h*)&xs[ 0 + bcol][k0 + bkb];
            v16h bf1 = *(const v16h*)&xs[16 + bcol][k0 + bkb];
            v16h bf2 = *(const v16h*)&xs[32 + bcol][k0 + bkb];
            v16h bf3 = *(const v16h*)&xs[48 + bcol][k0 + bkb];
            acc0 = __builtin_amdgcn_wmma_f32_16x16x32_f16(false, a.v, false, bf0, (short)0, acc0, false, false);
            acc1 = __builtin_amdgcn_wmma_f32_16x16x32_f16(false, a.v, false, bf1, (short)0, acc1, false, false);
            acc2 = __builtin_amdgcn_wmma_f32_16x16x32_f16(false, a.v, false, bf2, (short)0, acc2, false, false);
            acc3 = __builtin_amdgcn_wmma_f32_16x16x32_f16(false, a.v, false, bf3, (short)0, acc3, false, false);
        }

        // ---- epilogue: scale by invnorm[t], max/argmax over 64 t, merge ----
        // C layout: VGPR r, lanes 0-15 -> (M=r, N=lane); lanes 16-31 -> (M=r+8, N=lane-16)
        #pragma unroll
        for (int r = 0; r < 8; ++r) {
            const int t0r = t0 + tl;
            unsigned long long bk = 0ull;
            {
                int tt = t0r +  0; unsigned long long k = (tt < Tn) ? pack_key(acc0[r] * in0, tt) : 0ull; bk = k > bk ? k : bk;
                tt     = t0r + 16;                   k = (tt < Tn) ? pack_key(acc1[r] * in1, tt) : 0ull; bk = k > bk ? k : bk;
                tt     = t0r + 32;                   k = (tt < Tn) ? pack_key(acc2[r] * in2, tt) : 0ull; bk = k > bk ? k : bk;
                tt     = t0r + 48;                   k = (tt < Tn) ? pack_key(acc3[r] * in3, tt) : 0ull; bk = k > bk ? k : bk;
            }
            #pragma unroll
            for (int sh = 1; sh < 16; sh <<= 1) {    // reduce within each 16-lane half
                unsigned long long o = shfl_xor_u64(bk, sh);
                bk = o > bk ? o : bk;
            }
            if ((lane & 15) == 0) {
                int m = mw + r + ((lane >> 4) << 3); // lane 0 -> M=r, lane 16 -> M=r+8
                atomicMax(&best[(size_t)b * Mn + m], bk);   // GLOBAL_ATOMIC_MAX_U64
            }
        }
    }
}

// ---------- kernel 3: decode argmax table, thresholds, centroids ----------
__global__ __launch_bounds__(128) void finalize_kernel(const unsigned long long* __restrict__ best,
                                                       const float* __restrict__ thr,
                                                       const float* __restrict__ agg,
                                                       const int* __restrict__ p_stride,
                                                       const int* __restrict__ p_patch,
                                                       float* __restrict__ out) {
    const int c = blockIdx.x;   // class
    const int b = blockIdx.y;   // batch
    const int o = threadIdx.x;  // descriptor within class

    __shared__ float sx[TOn];
    __shared__ float sy[TOn];
    __shared__ int   sc[TOn];

    unsigned long long key = best[(size_t)b * Mn + c * TOn + o];
    unsigned s = (unsigned)(key >> 32);
    unsigned u = (s & 0x80000000u) ? (s & 0x7FFFFFFFu) : ~s;
    float sim  = __uint_as_float(u);
    int   tIdx = 0x7FFFFFFF - (int)(unsigned)(key & 0xFFFFFFFFu);

    bool found = sim > thr[c];
    int stride = *p_stride;
    int half   = (*p_patch) >> 1;
    int yd = tIdx / Pn;
    int xd = tIdx - yd * Pn;
    float xc = (float)((xd - 1) * stride + stride + half) - 0.5f;
    float yc = (float)((yd - 1) * stride + stride + half) - 0.5f;

    sx[o] = found ? xc : 0.0f;
    sy[o] = found ? yc : 0.0f;
    sc[o] = found ? 1 : 0;
    __syncthreads();
    #pragma unroll
    for (int step = TOn / 2; step > 0; step >>= 1) {
        if (o < step) { sx[o] += sx[o + step]; sy[o] += sy[o + step]; sc[o] += sc[o + step]; }
        __syncthreads();
    }
    if (o == 0) {
        int count  = sc[0];
        bool mask  = (float)count >= (float)TOn * agg[c];
        float cf   = fmaxf((float)count, 1.0f);
        float lx   = sx[0] / cf;
        float ly   = sy[0] / cf;
        float z    = (float)count / (float)TOn;
        float* op  = out + (size_t)b * (Cc * 3) + c * 3;
        op[0] = mask ? lx : 0.0f;
        op[1] = mask ? ly : 0.0f;
        op[2] = mask ? z  : 0.0f;
    }
}

extern "C" void kernel_launch(void* const* d_in, const int* in_sizes, int n_in,
                              void* d_out, int out_size, void* d_ws, size_t ws_size,
                              hipStream_t stream) {
    const float* x    = (const float*)d_in[0];
    const float* desc = (const float*)d_in[1];
    const float* thr  = (const float*)d_in[2];
    const float* agg  = (const float*)d_in[3];
    const int*   strd = (const int*)d_in[4];
    const int*   ptch = (const int*)d_in[5];
    float*       out  = (float*)d_out;

    // workspace layout: [0, 80KB) argmax table u64[B*M]; then f16 dn[M*D]
    unsigned long long* best = (unsigned long long*)d_ws;
    _Float16*           dn   = (_Float16*)((char*)d_ws + (size_t)Bn * Mn * sizeof(unsigned long long));

    const int nBest = Bn * Mn; // 10240
    init_best_kernel<<<(nBest + 255) / 256, 256, 0, stream>>>(best, nBest);
    prep_desc_kernel<<<Mn, 128, 0, stream>>>(desc, dn);
    dim3 grid(NTILES, Bn); // (186, 16) blocks x 256 threads; x read once per tile
    sim_argmax_kernel<<<grid, 256, 0, stream>>>(x, dn, best);
    finalize_kernel<<<dim3(Cc, Bn), TOn, 0, stream>>>(best, thr, agg, strd, ptch, out);
}
